// DMGI_34127810134588
// MI455X (gfx1250) — matrix operand order repaired and checked by reading, not compiled, not run on hardware
//
#include <hip/hip_runtime.h>
#include <stdint.h>

typedef __attribute__((ext_vector_type(16))) __bf16 v16bf;
typedef __attribute__((ext_vector_type(8)))  float  v8f;

#define FD 128   // feature dim D
#define FH 128   // hidden dim H

// ---------------------------------------------------------------------------
// helpers
// ---------------------------------------------------------------------------
static __device__ __forceinline__ uint32_t mix32(uint32_t a, uint32_t b) {
    uint32_t h = a * 0x9E3779B9u + b * 0x85EBCA6Bu;
    h ^= h >> 16; h *= 0x7FEB352Du;
    h ^= h >> 15; h *= 0x846CA68Bu;
    h ^= h >> 16;
    return h;
}

// native f32 -> bf16 convert (gfx1250 has hardware bf16 cvt)
static __device__ __forceinline__ __bf16 f2bf(float f) { return (__bf16)f; }

// ---------------------------------------------------------------------------
// zero the agg / cnt scratch (graph-capture safe, no hipMemset)
// ---------------------------------------------------------------------------
__global__ void dmgi_zero_kernel(float* __restrict__ p, long long n) {
    long long i = (long long)blockIdx.x * blockDim.x + threadIdx.x;
    long long stride = (long long)gridDim.x * blockDim.x;
    for (; i < n; i += stride) p[i] = 0.0f;
}

// ---------------------------------------------------------------------------
// pre-pack one [128,128] f32 weight matrix into bf16 WMMA B-fragment layout:
// pk[(((w*4 + kb)*32 + lane)*16 + j] = W[(kb*32 + (lane>>4)*16 + j)*FH + w*16 + (lane&15)]
// -> per (wave,kb) each lane's 16 B elements are contiguous (32B b128x2 load)
// ---------------------------------------------------------------------------
__global__ void dmgi_pack_w_kernel(const float* __restrict__ W,
                                   __bf16* __restrict__ pk) {
    int idx = blockIdx.x * blockDim.x + threadIdx.x;   // 0 .. 16384
    if (idx >= FD * FH) return;
    int j    = idx & 15;
    int lane = (idx >> 4) & 31;
    int kb   = (idx >> 9) & 3;
    int w    = idx >> 11;
    int k = kb * 32 + (lane >> 4) * 16 + j;
    int h = w * 16 + (lane & 15);
    pk[idx] = f2bf(W[(size_t)k * FH + h]);
}

// ---------------------------------------------------------------------------
// dropout (+ optional row permutation for the negative sample)
// neg = permutation(dropout(x)) => xd[i] = mask(p(i)) ? 2*x[p(i)] : 0
// ---------------------------------------------------------------------------
__global__ void dmgi_dropout_kernel(const float* __restrict__ x,
                                    float* __restrict__ xd,
                                    int N, uint32_t seed,
                                    int permute, uint32_t permC) {
    int total = N * FD;
    int i = blockIdx.x * blockDim.x + threadIdx.x;
    int stride = gridDim.x * blockDim.x;
    for (; i < total; i += stride) {
        int row = i >> 7;            // FD == 128
        int col = i & (FD - 1);
        int srow = row;
        if (permute)
            srow = (int)((((unsigned long long)row * 48271ull) + permC) % (uint32_t)N);
        int sidx = srow * FD + col;
        uint32_t h = mix32(seed, (uint32_t)sidx);
        float v = x[sidx];
        xd[i] = (h & 0x80000000u) ? v * 2.0f : 0.0f;   // keep-prob 0.5, scale 1/(1-p)
    }
}

// ---------------------------------------------------------------------------
// edge scatter: agg[dst] += xd[src]  (4 channels / thread), cnt[dst] += 1
// global_atomic_add_f32, L2-resident (agg = 25.6MB << 192MB L2)
// ---------------------------------------------------------------------------
__global__ void dmgi_scatter_kernel(const float* __restrict__ xd,
                                    const int* __restrict__ src,
                                    const int* __restrict__ dst,
                                    float* __restrict__ agg,
                                    float* __restrict__ cnt,
                                    int E) {
    long long total = (long long)E * (FD / 4);
    long long i = (long long)blockIdx.x * blockDim.x + threadIdx.x;
    long long stride = (long long)gridDim.x * blockDim.x;
    for (; i < total; i += stride) {
        int e  = (int)(i >> 5);         // FD/4 == 32 quads per edge
        int q  = (int)(i & 31);
        int d0 = q * 4;
        int s = src[e];
        int t = dst[e];
        const float4 v = *(const float4*)(xd + (size_t)s * FD + d0);
        float* ap = agg + (size_t)t * FD + d0;
        unsafeAtomicAdd(ap + 0, v.x);
        unsafeAtomicAdd(ap + 1, v.y);
        unsafeAtomicAdd(ap + 2, v.z);
        unsafeAtomicAdd(ap + 3, v.w);
        if (q == 0) unsafeAtomicAdd(&cnt[t], 1.0f);
    }
}

// ---------------------------------------------------------------------------
// fused: out = relu( (agg/max(cnt,1)) @ W_l + xd @ W_r + b_l )
// one wave -> 16x16 f32 tile via v_wmma_f32_16x16x32_bf16, 8 waves cover H=128
// A staged in LDS (bf16), B pre-packed bf16 fragments streamed from global/L2
// ---------------------------------------------------------------------------
__global__ __launch_bounds__(256)
void dmgi_sage_gemm_kernel(const float* __restrict__ agg,
                           const float* __restrict__ cnt,
                           const float* __restrict__ xd,
                           const __bf16* __restrict__ Wlpk,
                           const float* __restrict__ bl,
                           const __bf16* __restrict__ Wrpk,
                           float* __restrict__ out,
                           int N) {
    // 136-element row stride: row m starts at bank (4*m)%64 -> conflict-free
    __shared__ __bf16 As[16][136];   // mean-aggregated neighbors (bf16)
    __shared__ __bf16 Xs[16][136];   // dropped self features (bf16)

    int tile = blockIdx.x;
    int t = threadIdx.x;

    // stage A operands into LDS: float4 loads + mean normalize + bf16 convert
    for (int idx = t; idx < 16 * (FD / 4); idx += 256) {
        int row = idx >> 5;
        int col = (idx & 31) * 4;
        int node = tile * 16 + row;
        float4 a = make_float4(0.f, 0.f, 0.f, 0.f);
        float4 xv = a;
        if (node < N) {
            float c = cnt[node];
            float s = 1.0f / fmaxf(c, 1.0f);
            a = *(const float4*)(agg + (size_t)node * FD + col);
            a.x *= s; a.y *= s; a.z *= s; a.w *= s;
            xv = *(const float4*)(xd + (size_t)node * FD + col);
        }
        As[row][col + 0] = f2bf(a.x);
        As[row][col + 1] = f2bf(a.y);
        As[row][col + 2] = f2bf(a.z);
        As[row][col + 3] = f2bf(a.w);
        Xs[row][col + 0] = f2bf(xv.x);
        Xs[row][col + 1] = f2bf(xv.y);
        Xs[row][col + 2] = f2bf(xv.z);
        Xs[row][col + 3] = f2bf(xv.w);
    }
    __syncthreads();

    int lane  = t & 31;
    int wave  = t >> 5;
    int l15   = lane & 15;    // A row / B column within tile
    int khalf = lane >> 4;    // which K half this lane holds
    int hbase = wave * 16;    // 8 waves x 16 cols = H = 128

    const v16bf* Bl = (const v16bf*)Wlpk;   // [ (wave*4+kb)*32 + lane ]
    const v16bf* Br = (const v16bf*)Wrpk;

    v8f c = {};

    // ---- agg_mean @ W_l : K = 128 = 4 x 32 ----
#pragma unroll
    for (int kb = 0; kb < 4; ++kb) {
        v16bf a;
        int ka = kb * 32 + khalf * 16;
#pragma unroll
        for (int j = 0; j < 8; ++j) {
            a[j]     = As[l15][ka + j];        // K = kb*32 + khalf*16 + j
            a[8 + j] = As[l15][ka + 8 + j];    // K = kb*32 + khalf*16 + 8 + j
        }
        v16bf b = Bl[(wave * 4 + kb) * 32 + lane];
        c = __builtin_amdgcn_wmma_f32_16x16x32_bf16(false, a, false, b,
                                                    (short)0, c, false, false);
    }

    // ---- xd @ W_r : accumulate into same C ----
#pragma unroll
    for (int kb = 0; kb < 4; ++kb) {
        v16bf a;
        int ka = kb * 32 + khalf * 16;
#pragma unroll
        for (int j = 0; j < 8; ++j) {
            a[j]     = Xs[l15][ka + j];
            a[8 + j] = Xs[l15][ka + 8 + j];
        }
        v16bf b = Br[(wave * 4 + kb) * 32 + lane];
        c = __builtin_amdgcn_wmma_f32_16x16x32_bf16(false, a, false, b,
                                                    (short)0, c, false, false);
    }

    // bias + relu + store (C/D layout: lane(l15)=col, VGPR v -> row khalf*8+v)
    float bias = bl[hbase + l15];
    int m0 = khalf * 8;
#pragma unroll
    for (int v = 0; v < 8; ++v) {
        int node = tile * 16 + m0 + v;
        if (node < N) {
            float val = c[v] + bias;
            out[(size_t)node * FH + hbase + l15] = fmaxf(val, 0.0f);
        }
    }
}

// ---------------------------------------------------------------------------
// launcher
// ---------------------------------------------------------------------------
extern "C" void kernel_launch(void* const* d_in, const int* in_sizes, int n_in,
                              void* d_out, int out_size, void* d_ws, size_t ws_size,
                              hipStream_t stream) {
    (void)n_in; (void)out_size; (void)ws_size;

    const float* x  = (const float*)d_in[0];   // [N, D]
    const int*   ei = (const int*)d_in[1];     // [R, 2, E]
    const float* Wl = (const float*)d_in[2];   // [R, D, H]
    const float* bl = (const float*)d_in[3];   // [R, H]
    const float* Wr = (const float*)d_in[4];   // [R, D, H]
    float* out = (float*)d_out;                // pos [R,N,H] ++ neg [R,N,H]

    const int N = in_sizes[0] / FD;
    const int R = in_sizes[3] / FH;
    const int E = in_sizes[1] / (R * 2);

    // workspace layout: xd[N*D] f32 | agg[N*D] f32 | cnt[N] f32 | Wlpk | Wrpk (bf16)
    float* xd  = (float*)d_ws;
    float* agg = xd + (size_t)N * FD;
    float* cnt = agg + (size_t)N * FD;
    size_t cntPad = ((size_t)N + 63) & ~(size_t)63;      // keep bf16 buffers 32B-aligned
    __bf16* Wlpk = (__bf16*)(cnt + cntPad);
    __bf16* Wrpk = Wlpk + (size_t)FD * FH;

    const long long zn = (long long)N * FD + N;          // agg + cnt contiguous
    const int zGrid = (int)((zn + 255) / 256);
    const int dGrid = (N * FD + 255) / 256;
    const long long sTot = (long long)E * (FD / 4);
    const int sGrid = (int)((sTot + 255) / 256);
    const int tiles = (N + 15) / 16;
    const int pGrid = (FD * FH + 255) / 256;

    for (int r = 0; r < R; ++r) {
        const int* srcp = ei + (size_t)r * 2 * E;
        const int* dstp = srcp + E;
        const float* blr = bl + (size_t)r * FH;

        // pre-pack this relation's weights into bf16 WMMA fragment layout
        dmgi_pack_w_kernel<<<pGrid, 256, 0, stream>>>(Wl + (size_t)r * FD * FH, Wlpk);
        dmgi_pack_w_kernel<<<pGrid, 256, 0, stream>>>(Wr + (size_t)r * FD * FH, Wrpk);

        for (int pv = 0; pv < 2; ++pv) {                 // 0 = pos, 1 = neg
            uint32_t seed  = (uint32_t)(r * 2 + pv + 1) * 0x9E3779B9u + 0xC0FFEEu;
            uint32_t permC = (uint32_t)((r * 7919 + 104729) % N);

            dmgi_zero_kernel<<<zGrid, 256, 0, stream>>>(agg, zn);
            dmgi_dropout_kernel<<<dGrid, 256, 0, stream>>>(x, xd, N, seed, pv, permC);
            dmgi_scatter_kernel<<<sGrid, 256, 0, stream>>>(xd, srcp, dstp, agg, cnt, E);
            dmgi_sage_gemm_kernel<<<tiles, 256, 0, stream>>>(
                agg, cnt, xd, Wlpk, blr, Wrpk,
                out + (size_t)(pv * R + r) * N * FH, N);
        }
    }
}
